// AttentionLayer_84842783965222
// MI455X (gfx1250) — matrix operand order repaired and checked by reading, not compiled, not run on hardware
//
#include <hip/hip_runtime.h>
#include <stdint.h>

// ---------------------------------------------------------------------------
// Luong dot attention, flash style, bf16 WMMA (v_wmma_f32_16x16x32_bf16)
// B=16, Te=Td=2048, H=1024.  out[b,m,:] = concat(Dec[b,m,:], softmax(Dec.Enc^T).Enc)
//
// Per workgroup (8 wave32): one 16-row Td tile.
//  - GEMM1 (S = D.E^T): K=H split across 8 waves (128 each), B frags converted
//    f32->bf16 straight from global (E is L2-resident: 128 MB < 192 MB L2),
//    partials reduced with ds_add_f32.
//  - Online softmax per lane (row = lane&15), P A-fragment built in registers.
//  - GEMM2 (O += P.E): each wave owns a 128-wide H slice; B frags from a
//    pair-packed bf16 LDS buffer (stride 80 B -> conflict-free ds_load_b128).
// ---------------------------------------------------------------------------

typedef __attribute__((ext_vector_type(16))) __bf16 v16bf;
typedef __attribute__((ext_vector_type(8)))  float  v8f;

static constexpr int Bb  = 16;
static constexpr int Te  = 2048;
static constexpr int Td  = 2048;
static constexpr int Hh  = 1024;
static constexpr int MT  = 16;     // Td rows per workgroup
static constexpr int TET = 32;     // Te tile per iteration
static constexpr int BLOCK = 256;  // 8 wave32

// LDS layout (bytes), all strides padded for bank-conflict-free b128 access
static constexpr uint32_t D1_STRIDE = 2064;               // 1032 bf16 per row (16 rows)
static constexpr uint32_t D1_BYTES  = 16 * D1_STRIDE;     // 33024
static constexpr uint32_t E2_STRIDE = 80;                 // 16 pair-packed dwords + pad
static constexpr uint32_t E2_BYTES  = (uint32_t)Hh * E2_STRIDE; // 81920
static constexpr uint32_t S_STRIDE  = 144;                // 36 dwords per score row
static constexpr uint32_t S_BYTES   = 16 * S_STRIDE;      // 2304
static constexpr uint32_t SC_BYTES  = 8 * 16 * 4;         // per-wave row stats
static constexpr uint32_t LDS_TOTAL = D1_BYTES + E2_BYTES + S_BYTES + SC_BYTES;

union FragBF {
    uint32_t u[8];
    uint4    q[2];
    v16bf    v;
};

__device__ __forceinline__ uint32_t pack_bf16(float lo, float hi) {
    uint32_t l = __builtin_bit_cast(uint32_t, lo);
    uint32_t h = __builtin_bit_cast(uint32_t, hi);
    return (h & 0xFFFF0000u) | (l >> 16);   // -> single v_perm_b32
}

__global__ __launch_bounds__(BLOCK)
void luong_flash_attn(const float* __restrict__ Enc,
                      const float* __restrict__ Dec,
                      float* __restrict__ Out) {
    extern __shared__ __align__(16) unsigned char smem[];
    unsigned char* D1  = smem;                    // bf16 D tile, A-frag friendly
    unsigned char* E2  = smem + D1_BYTES;         // pair-packed bf16 E tile (GEMM2 B)
    unsigned char* Ssm = E2 + E2_BYTES;           // f32 score tile 16 x 32
    float*         SC  = (float*)(Ssm + S_BYTES); // per-wave row scale / l

    const int tid  = threadIdx.x;
    const int w    = tid >> 5;        // wave id 0..7  (owns H slice [128w,128w+128))
    const int lane = tid & 31;
    const int lh   = lane & 15;
    const int ahi  = lane >> 4;       // 0: lanes 0-15, 1: lanes 16-31

    const int b     = blockIdx.y;
    const int mBase = blockIdx.x * MT;

    const float* Eb = Enc + (size_t)b * Te * Hh;
    const float* Db = Dec + ((size_t)b * Td + mBase) * Hh;
    float*       Ob = Out + ((size_t)b * Td + mBase) * (size_t)(2 * Hh);

    // ---- one-time: copy decoder rows to Out[:, 0:H] and stage D tile as bf16 in LDS
    {
        const int r  = tid >> 4;      // 16 rows
        const int cc = tid & 15;      // 16 chunks of 64 cols
        const float4* src = (const float4*)(Db + (size_t)r * Hh + cc * 64);
        float4*       dst = (float4*)(Ob + (size_t)r * (2 * Hh) + cc * 64);
        uint32_t pk[32];
#pragma unroll
        for (int i = 0; i < 16; ++i) {
            float4 v = src[i];
            dst[i] = v;
            pk[2 * i]     = pack_bf16(v.x, v.y);
            pk[2 * i + 1] = pack_bf16(v.z, v.w);
        }
        uint4* d1p = (uint4*)(D1 + (uint32_t)r * D1_STRIDE + (uint32_t)cc * 128);
#pragma unroll
        for (int i = 0; i < 8; ++i)
            d1p[i] = make_uint4(pk[4 * i], pk[4 * i + 1], pk[4 * i + 2], pk[4 * i + 3]);
    }

    v8f O[8];
#pragma unroll
    for (int i = 0; i < 8; ++i) O[i] = (v8f){};
    float m_run = -__builtin_inff();
    float l_run = 0.0f;

    __syncthreads();   // D1 ready

    const int rq     = tid & 7;   // staging: row quad (rows 4rq..4rq+3 of tile)
    const int cchunk = tid >> 3;  // staging: 32-col chunk

    // each lane's 8 accumulator rows are v2 + 8*ahi -> its stats live at SC[w*16 + 8*ahi .. +8)
    const float4* myStat = (const float4*)(SC + w * 16 + 8 * ahi);

    for (int tB = 0; tB < Te; tB += TET) {
        __syncthreads();          // previous iteration consumers of E2/S done

        // ---- stage E2 (pair-packed bf16) and zero S
        {
            const float* er0 = Eb + (size_t)(tB + 4 * rq) * Hh;
            if (tB + TET < Te)
                __builtin_prefetch(er0 + (size_t)TET * Hh + cchunk * 32, 0, 1);
#pragma unroll
            for (int g = 0; g < 8; ++g) {
                const int c = cchunk * 32 + g * 4;
                float4 r0 = *(const float4*)(er0 + 0 * Hh + c);
                float4 r1 = *(const float4*)(er0 + 1 * Hh + c);
                float4 r2 = *(const float4*)(er0 + 2 * Hh + c);
                float4 r3 = *(const float4*)(er0 + 3 * Hh + c);
                *(uint2*)(E2 + (uint32_t)(c + 0) * E2_STRIDE + rq * 8) =
                    make_uint2(pack_bf16(r0.x, r1.x), pack_bf16(r2.x, r3.x));
                *(uint2*)(E2 + (uint32_t)(c + 1) * E2_STRIDE + rq * 8) =
                    make_uint2(pack_bf16(r0.y, r1.y), pack_bf16(r2.y, r3.y));
                *(uint2*)(E2 + (uint32_t)(c + 2) * E2_STRIDE + rq * 8) =
                    make_uint2(pack_bf16(r0.z, r1.z), pack_bf16(r2.z, r3.z));
                *(uint2*)(E2 + (uint32_t)(c + 3) * E2_STRIDE + rq * 8) =
                    make_uint2(pack_bf16(r0.w, r1.w), pack_bf16(r2.w, r3.w));
            }
            for (uint32_t o = (uint32_t)tid * 4; o < S_BYTES; o += BLOCK * 4)
                *(float*)(Ssm + o) = 0.0f;
        }

        // ---- GEMM1 partials: wave w contracts h in [128w, 128w+128)
        v8f S0 = (v8f){}, S1 = (v8f){};
        {
            const uint32_t kwave = (uint32_t)w * 128;
#pragma unroll
            for (int ks = 0; ks < 4; ++ks) {
                const uint32_t kbase = kwave + (uint32_t)ks * 32;
                // A fragment (D tile) from LDS: elems 0..7 at K=kbase+8*ahi, 8..15 at +16
                FragBF a;
                const uint4* ap = (const uint4*)(D1 + (uint32_t)lh * D1_STRIDE +
                                                 (kbase + 8u * ahi) * 2u);
                a.q[0] = ap[0];
                a.q[1] = ap[2];   // +32 bytes
                // B fragments (= E^T) straight from global / L2, convert f32->bf16
                const int hb = (int)kbase + 8 * ahi;
                const float4* bp0 = (const float4*)(Eb + (size_t)(tB + lh) * Hh + hb);
                const float4* bp1 = (const float4*)(Eb + (size_t)(tB + 16 + lh) * Hh + hb);
                FragBF fb0, fb1;
                {
                    float4 x0 = bp0[0], x1 = bp0[1], x2 = bp0[4], x3 = bp0[5];
                    fb0.u[0] = pack_bf16(x0.x, x0.y); fb0.u[1] = pack_bf16(x0.z, x0.w);
                    fb0.u[2] = pack_bf16(x1.x, x1.y); fb0.u[3] = pack_bf16(x1.z, x1.w);
                    fb0.u[4] = pack_bf16(x2.x, x2.y); fb0.u[5] = pack_bf16(x2.z, x2.w);
                    fb0.u[6] = pack_bf16(x3.x, x3.y); fb0.u[7] = pack_bf16(x3.z, x3.w);
                }
                {
                    float4 x0 = bp1[0], x1 = bp1[1], x2 = bp1[4], x3 = bp1[5];
                    fb1.u[0] = pack_bf16(x0.x, x0.y); fb1.u[1] = pack_bf16(x0.z, x0.w);
                    fb1.u[2] = pack_bf16(x1.x, x1.y); fb1.u[3] = pack_bf16(x1.z, x1.w);
                    fb1.u[4] = pack_bf16(x2.x, x2.y); fb1.u[5] = pack_bf16(x2.z, x2.w);
                    fb1.u[6] = pack_bf16(x3.x, x3.y); fb1.u[7] = pack_bf16(x3.z, x3.w);
                }
                S0 = __builtin_amdgcn_wmma_f32_16x16x32_bf16(false, a.v, false, fb0.v,
                                                             (short)0, S0, false, false);
                S1 = __builtin_amdgcn_wmma_f32_16x16x32_bf16(false, a.v, false, fb1.v,
                                                             (short)0, S1, false, false);
            }
        }

        __syncthreads();          // E2 staged + S zeroed
        // ---- reduce partial scores across waves (f32 LDS atomics)
        {
            unsigned char* sb = Ssm + (uint32_t)lh * 4u + (uint32_t)ahi * (8u * S_STRIDE);
#pragma unroll
            for (int v2 = 0; v2 < 8; ++v2) {
                atomicAdd((float*)(sb + (uint32_t)v2 * S_STRIDE), S0[v2]);
                atomicAdd((float*)(sb + (uint32_t)v2 * S_STRIDE + 64u), S1[v2]);
            }
        }
        __syncthreads();          // S complete

        // ---- online softmax, per lane (row = lh; both half-waves duplicate stats)
        float p[32];
        {
            const float4* srow = (const float4*)(Ssm + (uint32_t)lh * S_STRIDE);
            float s[32];
#pragma unroll
            for (int i = 0; i < 8; ++i) {
                float4 q = srow[i];
                s[4 * i] = q.x; s[4 * i + 1] = q.y; s[4 * i + 2] = q.z; s[4 * i + 3] = q.w;
            }
            float mloc = s[0];
#pragma unroll
            for (int i = 1; i < 32; ++i) mloc = fmaxf(mloc, s[i]);
            const float m_new = fmaxf(m_run, mloc);
            float sum = 0.0f;
#pragma unroll
            for (int i = 0; i < 32; ++i) { p[i] = __expf(s[i] - m_new); sum += p[i]; }
            const float scale = __expf(m_run - m_new);
            l_run = l_run * scale + sum;
            m_run = m_new;
            if (!ahi) SC[w * 16 + lh] = scale;   // per-wave stats region, no races
        }

        // ---- P fragment (A of GEMM2) built in registers, contiguous k = 16*ahi + s
        FragBF pf;
        {
            const int k0 = 16 * ahi;
#pragma unroll
            for (int i = 0; i < 4; ++i) pf.u[i]     = pack_bf16(p[k0 + 2 * i],      p[k0 + 2 * i + 1]);
#pragma unroll
            for (int i = 0; i < 4; ++i) pf.u[4 + i] = pack_bf16(p[8 + k0 + 2 * i], p[8 + k0 + 2 * i + 1]);
        }

        // ---- rescale accumulators: lane only needs its own 8 rows' scales
        {
            float4 sa = myStat[0], sb = myStat[1];    // SC[w*16 + 8*ahi .. +8)
            const float sc8[8] = {sa.x, sa.y, sa.z, sa.w, sb.x, sb.y, sb.z, sb.w};
#pragma unroll
            for (int v2 = 0; v2 < 8; ++v2) {
                const float f = sc8[v2];
#pragma unroll
                for (int nt = 0; nt < 8; ++nt) O[nt][v2] *= f;
            }
        }

        // ---- GEMM2: O[16 x 128w-slice] += P(16x32) x E(32x128)
#pragma unroll
        for (int nt = 0; nt < 8; ++nt) {
            const uint32_t c = (uint32_t)(w * 128 + nt * 16 + lh);
            const uint4* bp = (const uint4*)(E2 + c * E2_STRIDE + (uint32_t)ahi * 32u);
            FragBF eb;
            eb.q[0] = bp[0];
            eb.q[1] = bp[1];
            O[nt] = __builtin_amdgcn_wmma_f32_16x16x32_bf16(false, pf.v, false, eb.v,
                                                            (short)0, O[nt], false, false);
        }
    }

    // ---- finalize: broadcast l per row (per-wave LDS), normalize, store context half
    if (!ahi) SC[w * 16 + lh] = l_run;
    {
        float4 la = myStat[0], lb = myStat[1];        // this lane's 8 row sums
        const float li8[8] = {la.x, la.y, la.z, la.w, lb.x, lb.y, lb.z, lb.w};
#pragma unroll
        for (int v2 = 0; v2 < 8; ++v2) {
            const int   row = v2 + 8 * ahi;
            const float inv = 1.0f / li8[v2];
            float* orow = Ob + (size_t)row * (2 * Hh) + Hh;
#pragma unroll
            for (int nt = 0; nt < 8; ++nt)
                orow[w * 128 + nt * 16 + lh] = O[nt][v2] * inv;
        }
    }
}

extern "C" void kernel_launch(void* const* d_in, const int* in_sizes, int n_in,
                              void* d_out, int out_size, void* d_ws, size_t ws_size,
                              hipStream_t stream) {
    (void)in_sizes; (void)n_in; (void)out_size; (void)d_ws; (void)ws_size;
    const float* Enc = (const float*)d_in[0];   // encoder_outputs [B,Te,H]
    const float* Dec = (const float*)d_in[1];   // decoder_outputs [B,Td,H]
    float*       Out = (float*)d_out;           // [B,Td,2H]
    dim3 grid(Td / MT, Bb);
    luong_flash_attn<<<grid, BLOCK, LDS_TOTAL, stream>>>(Enc, Dec, Out);
}
